// GAT_75161927680606
// MI455X (gfx1250) — matrix operand order repaired and checked by reading, not compiled, not run on hardware
//
#include <hip/hip_runtime.h>

// ---------------- problem constants (match reference) ----------------
#define DIMF   64
#define HEADS  4
#define CPH    16          // channels per head
#define LAYERS 3
#define GG     16          // graphs in batch
#define NCLSK  2
#define NEG_ATT 0.2f
#define NEG_ACT 0.01f
#define EPSBN   1e-5f

typedef __attribute__((ext_vector_type(16))) __bf16 v16bf;
typedef __attribute__((ext_vector_type(8)))  float  v8f;

#if defined(__has_builtin)
#if __has_builtin(__builtin_amdgcn_global_load_async_to_lds_b128) && \
    __has_builtin(__builtin_amdgcn_s_wait_asynccnt)
#define HAVE_ASYNC_LDS 1
#endif
#endif

// 16-byte payload type + address-space pointer types for the async-LDS builtin
typedef int b128v __attribute__((vector_size(16)));
typedef __attribute__((address_space(1))) b128v* g_b128_ptr;   // global (HIP __device__)
typedef __attribute__((address_space(3))) b128v* l_b128_ptr;   // LDS

// order-preserving float<->uint encoding for atomic segment-max
__device__ __forceinline__ unsigned f2ord(float f) {
    unsigned u = __float_as_uint(f);
    return (u & 0x80000000u) ? ~u : (u | 0x80000000u);
}
__device__ __forceinline__ float ord2f(unsigned u) {
    return (u & 0x80000000u) ? __uint_as_float(u ^ 0x80000000u)
                             : __uint_as_float(~u);
}
__device__ __forceinline__ float lrelu(float x, float s) { return x > 0.f ? x : s * x; }

// ---------------- generic zero ----------------
__global__ void k_zero(float* __restrict__ p, long n) {
    long i = (long)blockIdx.x * blockDim.x + threadIdx.x;
    long st = (long)gridDim.x * blockDim.x;
    for (; i < n; i += st) p[i] = 0.f;
}

// ---------------- nodes-per-graph count ----------------
__global__ void k_count(const int* __restrict__ batch, float* __restrict__ cnt, int n) {
    int i = blockIdx.x * blockDim.x + threadIdx.x;
    if (i < n) atomicAdd(&cnt[batch[i]], 1.0f);
}

// ---------------- pooled sums of a [N,64] feature map ----------------
__global__ void k_poolsum(const float* __restrict__ x, const int* __restrict__ batch,
                          float* __restrict__ pool, int n) {
    long i = (long)blockIdx.x * blockDim.x + threadIdx.x;
    if (i >= (long)n * DIMF) return;
    int nid = (int)(i >> 6), c = (int)(i & 63);
    atomicAdd(&pool[batch[nid] * DIMF + c], x[i]);
}

// ---------------- h = Xcur @ W  via bf16 WMMA (f32 accumulate) ----------------
// block = 128 threads = 4 wave32; each wave computes one 16-row strip of all 64 cols.
// W staged in LDS via async-LDS copy (ASYNCcnt) when the toolchain exposes it.
// Epilogue staged through padded LDS so each lane emits coalesced b128 stores.
#define STRIDE_ST 68   // 16B-aligned (68*4=272) and rotates LDS banks by 4 per row

__global__ void k_gemm_xw(const float* __restrict__ X, const float* __restrict__ Wl,
                          float* __restrict__ Hout, int nrows) {
    __shared__ float Wlds[DIMF * DIMF];             // 16 KB (f32)
    __shared__ float stage[4][16 * STRIDE_ST];      // 4 waves x 4.25 KB
    int tid = threadIdx.x;

#if HAVE_ASYNC_LDS
    {   // 16 KB: 128 threads x 16 B x 8 passes, tracked by ASYNCcnt
        g_b128_ptr g = (g_b128_ptr)(void*)(float*)Wl;   // cast away const, AS1
        l_b128_ptr l = (l_b128_ptr)Wlds;                // AS3
#pragma unroll
        for (int p = 0; p < 8; ++p)
            __builtin_amdgcn_global_load_async_to_lds_b128(g + tid + p * 128,
                                                           l + tid + p * 128, 0, 0);
        __builtin_amdgcn_s_wait_asynccnt(0);
    }
#else
    for (int i = tid; i < DIMF * DIMF; i += blockDim.x) Wlds[i] = Wl[i];
#endif
    __syncthreads();

    int wave = tid >> 5;
    int lane = tid & 31;
    int row0 = (blockIdx.x * 4 + wave) * 16;        // wave-uniform
    if (row0 >= nrows) return;                      // whole wave exits together

    int half = lane >> 4;                           // lane half selects K sub-block
    int m    = lane & 15;
    int row  = row0 + m;
    if (row >= nrows) row = nrows - 1;              // clamp (stores guarded below)

    // A fragments: 16x32 bf16 per WMMA; two fragments cover K=0..63
    v16bf a0, a1;
    const float* xp = X + (long)row * DIMF;
#pragma unroll
    for (int j = 0; j < 16; ++j) {
        int pair = j >> 1, off = j & 1;
        int kk = (pair < 4) ? (pair * 2) : (16 + (pair - 4) * 2);
        int k  = kk + off + half * 8;
        a0[j] = (__bf16)xp[k];
        a1[j] = (__bf16)xp[32 + k];
    }

    int nn  = lane & 15;                            // output column within tile
    int kb0 = half * 8;                             // lane half -> K {0..7,16..23} or {8..15,24..31}
    v8f acc[4];
#pragma unroll
    for (int nt = 0; nt < 4; ++nt) {
        v16bf b0, b1;
#pragma unroll
        for (int j = 0; j < 16; ++j) {
            b0[j] = (__bf16)Wlds[(kb0 + j + (j & 8)) * DIMF + nt * 16 + nn];
            b1[j] = (__bf16)Wlds[(32 + kb0 + j + (j & 8)) * DIMF + nt * 16 + nn];
        }
        v8f c = {0.f, 0.f, 0.f, 0.f, 0.f, 0.f, 0.f, 0.f};
        c = __builtin_amdgcn_wmma_f32_16x16x32_bf16(false, a0, false, b0, (short)0, c, false, false);
        acc[nt] = __builtin_amdgcn_wmma_f32_16x16x32_bf16(false, a1, false, b1, (short)0, c, false, false);
    }

    // stage the 16x64 strip (fragment layout -> row-major, bank-rotated)
    float* st = stage[wave];
#pragma unroll
    for (int nt = 0; nt < 4; ++nt)
#pragma unroll
        for (int i = 0; i < 8; ++i)
            st[(half * 8 + i) * STRIDE_ST + nt * 16 + nn] = acc[nt][i];
    // same-wave LDS ops are in-order: loads below see the stores above

    int orow = lane & 15;
    int cb   = (lane >> 4) * 32;
    if (row0 + orow < nrows) {
        float4* dp = (float4*)(Hout + (long)(row0 + orow) * DIMF + cb);
        const float* sp = st + orow * STRIDE_ST + cb;
#pragma unroll
        for (int q = 0; q < 8; ++q) dp[q] = *(const float4*)(sp + q * 4);
    }
}

// ---------------- per-node attention logits ----------------
__global__ void k_logits(const float* __restrict__ h, const float* __restrict__ a_src,
                         const float* __restrict__ a_dst, float* __restrict__ e_src,
                         float* __restrict__ e_dst, int n) {
    int i = blockIdx.x * blockDim.x + threadIdx.x;
    if (i >= n * HEADS) return;
    int nid = i >> 2, hh = i & 3;
    const float* hp = h + (long)nid * DIMF + hh * CPH;
    float s0 = 0.f, s1 = 0.f;
#pragma unroll
    for (int c = 0; c < CPH; ++c) {
        float v = hp[c];
        s0 += v * a_src[hh * CPH + c];
        s1 += v * a_dst[hh * CPH + c];
    }
    e_src[i] = s0; e_dst[i] = s1;
}

// ---------------- edge pass A: alpha + segment max ----------------
__global__ void k_edge_alpha(const int* __restrict__ src, const int* __restrict__ dst,
                             const float* __restrict__ e_src, const float* __restrict__ e_dst,
                             float* __restrict__ alpha, unsigned* __restrict__ amax, long EH) {
    long i = (long)blockIdx.x * blockDim.x + threadIdx.x;
    if (i >= EH) return;
    long e = i >> 2; int hh = (int)(i & 3);
    int s = src[e], d = dst[e];
    float a = lrelu(e_src[s * HEADS + hh] + e_dst[d * HEADS + hh], NEG_ATT);
    alpha[i] = a;
    atomicMax(&amax[d * HEADS + hh], f2ord(a));
}

// ---------------- edge pass B: exp + denominator ----------------
__global__ void k_edge_exp(const int* __restrict__ dst, float* __restrict__ alpha,
                           const unsigned* __restrict__ amax, float* __restrict__ den, long EH) {
    long i = (long)blockIdx.x * blockDim.x + threadIdx.x;
    if (i >= EH) return;
    long e = i >> 2; int hh = (int)(i & 3);
    int d = dst[e];
    float ex = __expf(alpha[i] - ord2f(amax[d * HEADS + hh]));
    alpha[i] = ex;
    atomicAdd(&den[d * HEADS + hh], ex);
}

// ---------------- edge pass C: weighted scatter aggregation ----------------
__global__ void k_edge_agg(const int* __restrict__ src, const int* __restrict__ dst,
                           const float* __restrict__ exb, const float* __restrict__ den,
                           const float* __restrict__ h, float* __restrict__ out, long EH) {
    long i = (long)blockIdx.x * blockDim.x + threadIdx.x;
    if (i >= EH) return;
    long e = i >> 2; int hh = (int)(i & 3);
    int s = src[e], d = dst[e];
    const float* hrow = h + (long)s * DIMF + hh * CPH;
    __builtin_prefetch(hrow, 0, 0);                 // global_prefetch_b8: gather in flight
    float att = exb[i] / (den[d * HEADS + hh] + 1e-16f);
    const float4* hp = (const float4*)hrow;
    float* op = out + (long)d * DIMF + hh * CPH;
#pragma unroll
    for (int q = 0; q < 4; ++q) {
        float4 v = hp[q];
        atomicAdd(op + 4 * q + 0, v.x * att);
        atomicAdd(op + 4 * q + 1, v.y * att);
        atomicAdd(op + 4 * q + 2, v.z * att);
        atomicAdd(op + 4 * q + 3, v.w * att);
    }
}

// ---------------- batch-norm statistics (per-channel sum / sumsq) ----------------
__global__ void k_bnstats(const float* __restrict__ h, float* __restrict__ sum,
                          float* __restrict__ sumsq, int n) {
    int c = threadIdx.x;                 // blockDim.x == 64 (one channel per thread)
    float s = 0.f, sq = 0.f;
    for (int r = blockIdx.x; r < n; r += gridDim.x) {
        float v = h[(long)r * DIMF + c];
        s += v; sq += v * v;
    }
    atomicAdd(&sum[c], s);
    atomicAdd(&sumsq[c], sq);
}

// ---------------- BN + GraphSizeNorm + LeakyReLU + pool + residual ----------------
// (bias is all-zeros and cancels exactly through BN mean subtraction)
__global__ void k_norm(const float* __restrict__ agg, const float* __restrict__ sum,
                       const float* __restrict__ sumsq, const float* __restrict__ gamma,
                       const float* __restrict__ beta, const float* __restrict__ cnt,
                       const int* __restrict__ batch, float* __restrict__ xcur,
                       float* __restrict__ pool, int n) {
    long i = (long)blockIdx.x * blockDim.x + threadIdx.x;
    if (i >= (long)n * DIMF) return;
    int nid = (int)(i >> 6), c = (int)(i & 63);
    float inv_n = 1.0f / (float)n;
    float mu  = sum[c] * inv_n;
    float var = sumsq[c] * inv_n - mu * mu;
    float y = (agg[i] - mu) * rsqrtf(var + EPSBN) * gamma[c] + beta[c];
    int g = batch[nid];
    y *= rsqrtf(cnt[g]);
    y = lrelu(y, NEG_ACT);
    atomicAdd(&pool[g * DIMF + c], y);
    xcur[i] += y;                        // residual for next layer
}

// ---------------- readout head: risk, softmax, argmax ----------------
__global__ void k_head(const float* __restrict__ pools, const float* __restrict__ cnt,
                       const float* __restrict__ Wr, const float* __restrict__ br,
                       float* __restrict__ out) {
    int g = threadIdx.x;
    if (g >= GG) return;
    float r0 = br[0], r1 = br[1];
    float inv_c = 1.0f / cnt[g];
    for (int l = 0; l <= LAYERS; ++l) {
        for (int c = 0; c < DIMF; ++c) {
            float v = pools[(l * GG + g) * DIMF + c] * inv_c;
            int j = l * DIMF + c;
            r0 += v * Wr[j * NCLSK + 0];
            r1 += v * Wr[j * NCLSK + 1];
        }
    }
    float m  = fmaxf(r0, r1);
    float e0 = __expf(r0 - m), e1 = __expf(r1 - m);
    float sd = e0 + e1;
    out[g * 2 + 0] = r0;                 // risk
    out[g * 2 + 1] = r1;
    out[GG * 2 + g * 2 + 0] = e0 / sd;   // Y_prob
    out[GG * 2 + g * 2 + 1] = e1 / sd;
    out[GG * 4 + g] = (r1 > r0) ? 1.f : 0.f;  // Y_hat
}

// =====================================================================
extern "C" void kernel_launch(void* const* d_in, const int* in_sizes, int n_in,
                              void* d_out, int out_size, void* d_ws, size_t ws_size,
                              hipStream_t stream) {
    const float* x        = (const float*)d_in[0];
    const int*   ei       = (const int*)  d_in[1];
    const int*   batch    = (const int*)  d_in[2];
    // d_in[3] = n_graphs scalar (GG compile-time)
    const float* W        = (const float*)d_in[4];   // [L,64,64]
    const float* att_src  = (const float*)d_in[5];   // [L,H,C]
    const float* att_dst  = (const float*)d_in[6];
    // d_in[7] = bias (zeros; cancels through BN)
    const float* gamma    = (const float*)d_in[8];   // [L,64]
    const float* beta     = (const float*)d_in[9];
    const float* Wr       = (const float*)d_in[10];  // [(L+1)*64, 2]
    const float* br       = (const float*)d_in[11];

    const int N = in_sizes[0] / DIMF;
    const int E = in_sizes[1] / 2;
    const int* src = ei;
    const int* dst = ei + E;

    const long NH = (long)N * HEADS;
    const long EH = (long)E * HEADS;
    const long ND = (long)N * DIMF;

    // ---- carve workspace ----
    char* w = (char*)d_ws;
    auto carve = [&](size_t bytes) { char* p = w; w += (bytes + 255) & ~(size_t)255; return p; };
    float*    xcur   = (float*)   carve((size_t)N * DIMF * 4);
    float*    hbuf   = (float*)   carve((size_t)N * DIMF * 4);
    float*    agg    = (float*)   carve((size_t)N * DIMF * 4);
    float*    esrc   = (float*)   carve((size_t)N * HEADS * 4);
    float*    edst   = (float*)   carve((size_t)N * HEADS * 4);
    float*    maxden = (float*)   carve((size_t)2 * NH * 4);   // [amax | den] contiguous
    unsigned* amax   = (unsigned*)maxden;
    float*    den    = maxden + NH;
    float*    exb    = (float*)   carve((size_t)E * HEADS * 4);
    float*    cnt    = (float*)   carve(GG * 4);
    float*    pools  = (float*)   carve((LAYERS + 1) * GG * DIMF * 4);
    float*    bnbuf  = (float*)   carve(2 * DIMF * 4);         // [sum | sumsq]
    float*    bns    = bnbuf;
    float*    bnsq   = bnbuf + DIMF;

    const int ZB = 2048;

    // working copy of node features (inputs must not be mutated)
    (void)hipMemcpyAsync(xcur, x, (size_t)N * DIMF * 4, hipMemcpyDeviceToDevice, stream);

    k_zero<<<1, 256, 0, stream>>>(cnt, GG);
    k_zero<<<16, 256, 0, stream>>>(pools, (LAYERS + 1) * GG * DIMF);
    k_count<<<(N + 255) / 256, 256, 0, stream>>>(batch, cnt, N);
    k_poolsum<<<(int)((ND + 255) / 256), 256, 0, stream>>>(x, batch, pools, N);

    const int tiles = (N + 15) / 16;
    const int gemm_blocks = (tiles + 3) / 4;

    for (int l = 0; l < LAYERS; ++l) {
        k_zero<<<ZB, 256, 0, stream>>>(maxden, 2 * NH);     // 0 encodes "< everything"
        k_zero<<<ZB, 256, 0, stream>>>(agg, ND);
        k_zero<<<1, 128, 0, stream>>>(bnbuf, 2 * DIMF);

        k_gemm_xw<<<gemm_blocks, 128, 0, stream>>>(xcur, W + (size_t)l * DIMF * DIMF, hbuf, N);
        k_logits<<<(int)((NH + 255) / 256), 256, 0, stream>>>(
            hbuf, att_src + l * HEADS * CPH, att_dst + l * HEADS * CPH, esrc, edst, N);
        k_edge_alpha<<<(int)((EH + 255) / 256), 256, 0, stream>>>(src, dst, esrc, edst, exb, amax, EH);
        k_edge_exp<<<(int)((EH + 255) / 256), 256, 0, stream>>>(dst, exb, amax, den, EH);
        k_edge_agg<<<(int)((EH + 255) / 256), 256, 0, stream>>>(src, dst, exb, den, hbuf, agg, EH);
        k_bnstats<<<512, 64, 0, stream>>>(agg, bns, bnsq, N);
        k_norm<<<(int)((ND + 255) / 256), 256, 0, stream>>>(
            agg, bns, bnsq, gamma + l * DIMF, beta + l * DIMF, cnt, batch, xcur,
            pools + (size_t)(l + 1) * GG * DIMF, N);
    }

    k_head<<<1, 32, 0, stream>>>(pools, cnt, Wr, br, (float*)d_out);
}
